// RNN_16552803959291
// MI455X (gfx1250) — matrix operand order repaired and checked by reading, not compiled, not run on hardware
//
#include <hip/hip_runtime.h>
#include <hip/hip_bf16.h>
#include <math.h>

// Problem constants (match reference)
constexpr int B = 16, T = 1024, D = 128, H = 1024, O = 64;
constexpr float ALPHA = 0.1f;
constexpr int NWG = 32;     // persistent workgroups for the scan (32 neurons each)
constexpr int BDIM = 256;   // 8 waves of 32

typedef __attribute__((ext_vector_type(16))) __bf16 v16bf;
typedef __attribute__((ext_vector_type(8)))  __bf16 v8bf;
typedef __attribute__((ext_vector_type(8)))  float  v8f;
typedef __attribute__((ext_vector_type(4)))  unsigned int u32x4;
typedef __attribute__((ext_vector_type(8)))  int i32x8;
typedef __attribute__((ext_vector_type(4)))  int i32x4;

// ---------------------------------------------------------------------------
// WMMA helper: D = A(16x32 bf16) * B(32x16 bf16) + C(16x16 f32)
// ---------------------------------------------------------------------------
__device__ __forceinline__ v8f wmma_bf16(v16bf a, v16bf b, v8f c) {
  return __builtin_amdgcn_wmma_f32_16x16x32_bf16(
      /*neg_a=*/false, a, /*neg_b=*/false, b,
      /*c_mod=*/(short)0, c, /*reuse_a=*/false, /*reuse_b=*/false);
}

// A fragment (16x32, bf16, row-major source, leading dim ld):
//   lanes 0-15 : row=lane,    K = {k0..k0+7}  and {k0+16..k0+23}
//   lanes16-31 : row=lane-16, K = {k0+8..+15} and {k0+24..+31}
__device__ __forceinline__ v16bf frag_a_bf16(const __bf16* base, int ld, int k0, int lane) {
  int row = lane & 15;
  int kk  = k0 + ((lane >> 4) << 3);
  const __bf16* p = base + (size_t)row * ld + kk;
  v8bf lo = *(const v8bf*)(p);
  v8bf hi = *(const v8bf*)(p + 16);
  v16bf r;
#pragma unroll
  for (int i = 0; i < 8; ++i) { r[i] = lo[i]; r[i + 8] = hi[i]; }
  return r;
}

// B fragment (32x16, bf16). Source is W[n][k] row-major (k contiguous):
//   lanes 0-15 : col=lane,    K = {k0..k0+7}  and {k0+16..k0+23}
//   lanes16-31 : col=lane-16, K = {k0+8..+15} and {k0+24..+31}
__device__ __forceinline__ v16bf frag_b_bf16(const __bf16* base, int ld, int k0, int lane) {
  int col = lane & 15;
  int kk  = k0 + ((lane >> 4) << 3);
  const __bf16* p = base + (size_t)col * ld + kk;
  v8bf lo = *(const v8bf*)(p);
  v8bf hi = *(const v8bf*)(p + 16);
  v16bf r;
#pragma unroll
  for (int i = 0; i < 8; ++i) { r[i] = lo[i]; r[i + 8] = hi[i]; }
  return r;
}

// ---------------------------------------------------------------------------
// TDM: stage a contiguous [nrows x rowlen] bf16 tile from global into LDS.
// D# built per CDNA5 ISA 8.3/8.4 (group0: count|lds|global|type=2;
// group1: data_size=2B, dims, stride). 6-arg toolchain form:
//   (u32x4 g0, i32x8 g1, i32x4 g2, i32x4 g3, i32x8 extra, i32 cpol)
// Falls back to a cooperative copy if the builtin is unavailable.
// ---------------------------------------------------------------------------
#if __has_builtin(__builtin_amdgcn_tensor_load_to_lds)
#define HAVE_TDM 1
__device__ __forceinline__ void tdm_load_2d(unsigned lds_addr, const void* gptr,
                                            unsigned rowlen, unsigned nrows) {
  unsigned long long ga = (unsigned long long)gptr;
  u32x4 g0;
  g0[0] = 1u;                                              // count=1 (valid user D#)
  g0[1] = lds_addr;                                        // lds_addr (bytes)
  g0[2] = (unsigned)(ga & 0xffffffffu);                    // global_addr[31:0]
  g0[3] = (unsigned)((ga >> 32) & 0x1ffffffu) | (2u << 30);// addr[56:32] | type=2
  i32x8 g1;
  g1[0] = 0x00010000;                                      // wg_mask=0, data_size=1 (2B)
  g1[1] = (int)((rowlen & 0xffffu) << 16);                 // tensor_dim0[15:0]
  g1[2] = (int)((rowlen >> 16) & 0xffffu) | (int)((nrows & 0xffffu) << 16);
  g1[3] = (int)((nrows >> 16) & 0xffffu) | (int)((rowlen & 0xffffu) << 16); // tile_dim0
  g1[4] = (int)(nrows & 0xffffu);                          // tile_dim1, tile_dim2=0
  g1[5] = (int)rowlen;                                     // tensor_dim0_stride[31:0]
  g1[6] = 0;
  g1[7] = 0;
  i32x4 z4; z4[0] = 0; z4[1] = 0; z4[2] = 0; z4[3] = 0;    // groups 2/3 unused (2-D)
  i32x8 z8;
#pragma unroll
  for (int i = 0; i < 8; ++i) z8[i] = 0;
  __builtin_amdgcn_tensor_load_to_lds(g0, g1, z4, z4, z8, 0);
}
#else
#define HAVE_TDM 0
#endif

// ---------------------------------------------------------------------------
// Grid barrier: CDNA5 cluster barrier fast-path (NOP if not cluster-dispatched)
// + monotonic atomic counter as the portable path.
// ---------------------------------------------------------------------------
__device__ __forceinline__ void grid_barrier(unsigned* ctr, unsigned target) {
  __threadfence();
  __syncthreads();
  __builtin_amdgcn_s_cluster_barrier();
  if (threadIdx.x == 0) {
    __hip_atomic_fetch_add(ctr, 1u, __ATOMIC_RELEASE, __HIP_MEMORY_SCOPE_AGENT);
    while (__hip_atomic_load(ctr, __ATOMIC_ACQUIRE, __HIP_MEMORY_SCOPE_AGENT) < target) {
      __builtin_amdgcn_s_sleep(2);
    }
  }
  __syncthreads();
}

// ---------------------------------------------------------------------------
// Kernel 1: one-time bf16 packing. W_rec0 -> [H][1024]; {W_in1|W_rec1} ->
// [H][2048] (layer-1 becomes one K=2048 GEMM); x, W_in0, W_out -> bf16.
// Also resets the barrier counter.
// ---------------------------------------------------------------------------
__global__ void pack_weights(const float* __restrict__ Wr0f,
                             const float* __restrict__ Wi1f,
                             const float* __restrict__ Wr1f,
                             const float* __restrict__ xf,
                             const float* __restrict__ Win0f,
                             const float* __restrict__ Woutf,
                             __bf16* __restrict__ Wr0, __bf16* __restrict__ Wl1,
                             __bf16* __restrict__ xb, __bf16* __restrict__ Win0b,
                             __bf16* __restrict__ Woutb,
                             unsigned* __restrict__ bar) {
  int i = blockIdx.x * blockDim.x + threadIdx.x;
  if (i == 0) *bar = 0u;
  if (i < H * H) {
    Wr0[i] = (__bf16)Wr0f[i];
    int n = i >> 10, k = i & 1023;
    Wl1[(size_t)n * 2048 + k]        = (__bf16)Wi1f[i];
    Wl1[(size_t)n * 2048 + 1024 + k] = (__bf16)Wr1f[i];
  }
  if (i < B * T * D) xb[i]  = (__bf16)xf[i];
  if (i < H * D)     Win0b[i] = (__bf16)Win0f[i];
  if (i < O * H)     Woutb[i] = (__bf16)Woutf[i];
}

// ---------------------------------------------------------------------------
// Kernel 2: U0[b,t,h] = x[b,t,:] @ W_in0[h,:] + b0[h]  (M=16384,N=1024,K=128)
// ---------------------------------------------------------------------------
__global__ void u0_gemm(const __bf16* __restrict__ xb, const __bf16* __restrict__ Win0b,
                        const float* __restrict__ b0, float* __restrict__ U0) {
  int gw   = blockIdx.x * (blockDim.x >> 5) + (threadIdx.x >> 5);
  int lane = threadIdx.x & 31;
  int mt = gw >> 6;       // 1024 M-tiles
  int nt = gw & 63;       // 64 N-tiles
  if (mt >= (B * T) / 16) return;
  const __bf16* A  = xb    + (size_t)(mt * 16) * D;
  const __bf16* Bw = Win0b + (size_t)(nt * 16) * D;
  v8f c;
#pragma unroll
  for (int i = 0; i < 8; ++i) c[i] = 0.f;
#pragma unroll
  for (int k0 = 0; k0 < D; k0 += 32)
    c = wmma_bf16(frag_a_bf16(A, D, k0, lane), frag_b_bf16(Bw, D, k0, lane), c);
  int n  = nt * 16 + (lane & 15);
  int m0 = mt * 16 + ((lane >> 4) << 3);
  float bb = b0[n];
#pragma unroll
  for (int r = 0; r < 8; ++r)
    U0[(size_t)(m0 + r) * H + n] = c[r] + bb;
}

// ---------------------------------------------------------------------------
// Kernel 3: persistent time scan. 32 WGs x 8 waves; WG owns 32 neurons of both
// layers. All recurrent weights for the WG (192 KB bf16) are TDM-staged into
// LDS once and reused for all 1024 steps. Each 16x16 (batch x neuron) tile is
// computed by 4 waves splitting K, reduced through LDS. Membrane state v stays
// in registers. fr vectors exchanged via bf16 double buffers + grid barriers.
// ---------------------------------------------------------------------------
__global__ void __launch_bounds__(BDIM) scan_kernel(
    const float* __restrict__ U0, const float* __restrict__ b1,
    const __bf16* __restrict__ Wr0, const __bf16* __restrict__ Wl1,
    __bf16* __restrict__ fr0buf, __bf16* __restrict__ fr1buf,
    float* __restrict__ s0out, float* __restrict__ s1out,
    __bf16* __restrict__ s1b, unsigned* __restrict__ bar) {
  extern __shared__ __bf16 smem[];
  __bf16* Wr0s = smem;                        // 32x1024 bf16 = 64 KB
  __bf16* Wl1s = smem + 32 * 1024;            // 32x2048 bf16 = 128 KB
  float*  red  = (float*)(smem + 32 * 1024 + 32 * 2048);   // 2*3*256 f32 = 6 KB

  const int wg   = blockIdx.x;        // 0..31
  const int tid  = threadIdx.x;
  const int lane = tid & 31;
  const int wave = tid >> 5;          // 0..7
  const int tile = wave >> 2;         // 0..1  (16 neurons each)
  const int kq   = wave & 3;          // K quarter
  const int nloc = tile * 16;         // local row base inside the LDS slice
  const int n    = wg * 32 + nloc + (lane & 15);
  const int mrow0 = (lane >> 4) << 3;

  // ---- stage this WG's weight slices into LDS (TDM; fallback: coop copy) ----
  const __bf16* gWr0s = Wr0 + (size_t)wg * 32 * 1024;   // contiguous 64 KB
  const __bf16* gWl1s = Wl1 + (size_t)wg * 32 * 2048;   // contiguous 128 KB
#if HAVE_TDM
  if (wave == 0) {
    unsigned l0 = (unsigned)(unsigned long long)(void*)Wr0s;
    unsigned l1 = (unsigned)(unsigned long long)(void*)Wl1s;
    tdm_load_2d(l0, gWr0s, 1024, 32);
    tdm_load_2d(l1, gWl1s, 2048, 32);
    __builtin_amdgcn_s_wait_tensorcnt(0);
  }
#else
  for (int i = tid; i < 32 * 1024 / 8; i += BDIM) ((v8bf*)Wr0s)[i] = ((const v8bf*)gWr0s)[i];
  for (int i = tid; i < 32 * 2048 / 8; i += BDIM) ((v8bf*)Wl1s)[i] = ((const v8bf*)gWl1s)[i];
#endif

  float v0s[8], v1s[8];
#pragma unroll
  for (int i = 0; i < 8; ++i) { v0s[i] = 0.f; v1s[i] = 0.f; }
  const float b1n = ALPHA * b1[n];

  // zero parity-0 fr buffers (t=0 reads them)
  for (int i = tid + wg * BDIM; i < B * H; i += NWG * BDIM) {
    fr0buf[i] = (__bf16)0.f;
    fr1buf[i] = (__bf16)0.f;
  }
  __syncthreads();
  unsigned barcnt = 0;
  grid_barrier(bar, ++barcnt * NWG);

  for (int t = 0; t < T; ++t) {
    const int p = t & 1;
    const __bf16* fr0o = fr0buf + (size_t)p * B * H;
    __bf16*       fr0n = fr0buf + (size_t)(1 - p) * B * H;
    const __bf16* fr1o = fr1buf + (size_t)p * B * H;
    __bf16*       fr1n = fr1buf + (size_t)(1 - p) * B * H;

    // ---- layer 0: C = fr0_old @ W_rec0_sliceT (K=1024, split 4 ways) ----
    v8f c;
#pragma unroll
    for (int i = 0; i < 8; ++i) c[i] = 0.f;
#pragma unroll
    for (int k0 = kq * 256; k0 < kq * 256 + 256; k0 += 32) {
      v16bf a = frag_a_bf16(fr0o, H, k0, lane);
      v16bf b = frag_b_bf16(Wr0s + (size_t)nloc * 1024, 1024, k0, lane);  // LDS
      c = wmma_bf16(a, b, c);
    }
    if (kq > 0) {
#pragma unroll
      for (int r = 0; r < 8; ++r) red[(tile * 3 + kq - 1) * 256 + lane * 8 + r] = c[r];
    }
    __syncthreads();
    if (kq == 0) {
#pragma unroll
      for (int r = 0; r < 8; ++r) {
        float acc = c[r] + red[(tile * 3 + 0) * 256 + lane * 8 + r]
                         + red[(tile * 3 + 1) * 256 + lane * 8 + r]
                         + red[(tile * 3 + 2) * 256 + lane * 8 + r];
        int m = mrow0 + r;
        float u = U0[((size_t)m * T + t) * H + n];   // b0 folded in
        if (t + 1 < T) __builtin_prefetch(&U0[((size_t)m * T + t + 1) * H + n], 0, 1);
        v0s[r] = (1.f - ALPHA) * v0s[r] + ALPHA * (u + acc);
        float f = tanhf(v0s[r]);
        fr0n[(size_t)m * H + n] = (__bf16)f;
        s0out[((size_t)m * T + t) * H + n] = f;
      }
    }
    grid_barrier(bar, ++barcnt * NWG);   // fr0_new visible to all WGs

    // ---- layer 1: C = [fr0_new | fr1_old] @ {W_in1|W_rec1}T (K=2048) ----
    v8f c1;
#pragma unroll
    for (int i = 0; i < 8; ++i) c1[i] = 0.f;
#pragma unroll
    for (int k0 = kq * 512; k0 < kq * 512 + 512; k0 += 32) {
      v16bf a = (k0 < H) ? frag_a_bf16(fr0n, H, k0, lane)
                         : frag_a_bf16(fr1o, H, k0 - H, lane);
      v16bf b = frag_b_bf16(Wl1s + (size_t)nloc * 2048, 2048, k0, lane);  // LDS
      c1 = wmma_bf16(a, b, c1);
    }
    if (kq > 0) {
#pragma unroll
      for (int r = 0; r < 8; ++r) red[(tile * 3 + kq - 1) * 256 + lane * 8 + r] = c1[r];
    }
    __syncthreads();
    if (kq == 0) {
#pragma unroll
      for (int r = 0; r < 8; ++r) {
        float acc = c1[r] + red[(tile * 3 + 0) * 256 + lane * 8 + r]
                          + red[(tile * 3 + 1) * 256 + lane * 8 + r]
                          + red[(tile * 3 + 2) * 256 + lane * 8 + r];
        int m = mrow0 + r;
        v1s[r] = (1.f - ALPHA) * v1s[r] + ALPHA * acc + b1n;
        float f = tanhf(v1s[r]);
        fr1n[(size_t)m * H + n] = (__bf16)f;
        s1out[((size_t)m * T + t) * H + n] = f;
        s1b[((size_t)m * T + t) * H + n]   = (__bf16)f;   // bf16 shadow for readout
      }
    }
    grid_barrier(bar, ++barcnt * NWG);   // fr1_new visible before next step
  }
}

// ---------------------------------------------------------------------------
// Kernel 4: readout  out = states1 @ W_outT + b_out  (M=16384,N=64,K=1024)
// ---------------------------------------------------------------------------
__global__ void out_gemm(const __bf16* __restrict__ s1b, const __bf16* __restrict__ Woutb,
                         const float* __restrict__ bout, float* __restrict__ out) {
  int gw   = blockIdx.x * (blockDim.x >> 5) + (threadIdx.x >> 5);
  int lane = threadIdx.x & 31;
  int mt = gw >> 2;        // 1024 M-tiles
  int nt = gw & 3;         // 4 N-tiles
  if (mt >= (B * T) / 16) return;
  const __bf16* A  = s1b   + (size_t)(mt * 16) * H;
  const __bf16* Bw = Woutb + (size_t)(nt * 16) * H;
  v8f c;
#pragma unroll
  for (int i = 0; i < 8; ++i) c[i] = 0.f;
  for (int k0 = 0; k0 < H; k0 += 32)
    c = wmma_bf16(frag_a_bf16(A, H, k0, lane), frag_b_bf16(Bw, H, k0, lane), c);
  int o  = nt * 16 + (lane & 15);
  int m0 = mt * 16 + ((lane >> 4) << 3);
  float bb = bout[o];
#pragma unroll
  for (int r = 0; r < 8; ++r)
    out[(size_t)(m0 + r) * O + o] = c[r] + bb;
}

// ---------------------------------------------------------------------------
extern "C" void kernel_launch(void* const* d_in, const int* in_sizes, int n_in,
                              void* d_out, int out_size, void* d_ws, size_t ws_size,
                              hipStream_t stream) {
  const float* x     = (const float*)d_in[0];
  const float* Win0  = (const float*)d_in[1];
  const float* Wrec0 = (const float*)d_in[2];
  const float* b0    = (const float*)d_in[3];
  const float* Win1  = (const float*)d_in[4];
  const float* Wrec1 = (const float*)d_in[5];
  const float* b1    = (const float*)d_in[6];
  const float* Wout  = (const float*)d_in[7];
  const float* bout  = (const float*)d_in[8];

  float* out = (float*)d_out;                 // [B,T,O]
  float* s0  = out + (size_t)B * T * O;       // [B,T,H]
  float* s1  = s0  + (size_t)B * T * H;       // [B,T,H]

  // workspace carve-out (~107 MB)
  char* w = (char*)d_ws;
  unsigned* bar  = (unsigned*)w;             w += 256;
  __bf16* fr0    = (__bf16*)w;               w += (size_t)2 * B * H * sizeof(__bf16);
  __bf16* fr1    = (__bf16*)w;               w += (size_t)2 * B * H * sizeof(__bf16);
  __bf16* Wr0b   = (__bf16*)w;               w += (size_t)H * H * sizeof(__bf16);
  __bf16* Wl1b   = (__bf16*)w;               w += (size_t)H * 2048 * sizeof(__bf16);
  __bf16* xb     = (__bf16*)w;               w += (size_t)B * T * D * sizeof(__bf16);
  __bf16* Win0b  = (__bf16*)w;               w += (size_t)H * D * sizeof(__bf16);
  __bf16* Woutb  = (__bf16*)w;               w += (size_t)O * H * sizeof(__bf16);
  __bf16* s1b    = (__bf16*)w;               w += (size_t)B * T * H * sizeof(__bf16);
  float*  U0     = (float*)w;                // [B*T*H] f32 (64 MB)

  const int packN = B * T * D;               // largest elementwise range (2M)
  pack_weights<<<(packN + 255) / 256, 256, 0, stream>>>(
      Wrec0, Win1, Wrec1, x, Win0, Wout, Wr0b, Wl1b, xb, Win0b, Woutb, bar);
  u0_gemm<<<(B * T / 16) * (H / 16) / 8, BDIM, 0, stream>>>(xb, Win0b, b0, U0);

  const size_t SCAN_LDS = (size_t)(32 * 1024 + 32 * 2048) * sizeof(__bf16)
                        + (size_t)2 * 3 * 256 * sizeof(float);   // 202,752 B
  (void)hipFuncSetAttribute((const void*)scan_kernel,
                            hipFuncAttributeMaxDynamicSharedMemorySize, (int)SCAN_LDS);
  scan_kernel<<<NWG, BDIM, SCAN_LDS, stream>>>(U0, b1, Wr0b, Wl1b, fr0, fr1,
                                               s0, s1, s1b, bar);
  out_gemm<<<(B * T / 16) * (O / 16) / 8, BDIM, 0, stream>>>(s1b, Woutb, bout, out);
}
